// Encoder_36258113913425
// MI455X (gfx1250) — matrix operand order repaired and checked by reading, not compiled, not run on hardware
//
#include <hip/hip_runtime.h>
#include <hip/hip_bf16.h>

typedef __bf16 bf16;
typedef bf16 v16bf __attribute__((ext_vector_type(16)));
typedef bf16 v8bf  __attribute__((ext_vector_type(8)));
typedef float v8f  __attribute__((ext_vector_type(8)));

union FragA { v16bf v; v8bf h[2]; };

__device__ __forceinline__ v8f vzero8f() {
    v8f z = {0.f, 0.f, 0.f, 0.f, 0.f, 0.f, 0.f, 0.f};
    return z;
}

__device__ __forceinline__ v8f wmma_bf16(v16bf a, v16bf b, v8f c) {
    // D(f32 16x16) = A(bf16 16x32) * B(bf16 32x16) + C
    return __builtin_amdgcn_wmma_f32_16x16x32_bf16(
        /*neg_a=*/false, a, /*neg_b=*/false, b,
        /*c_mod=*/(short)0, c, /*reuse_a=*/false, /*reuse_b=*/false);
}

// --- DPP16 butterfly reduction over each 16-lane half (pure VALU, no LDS) ---
template <int CTRL>
__device__ __forceinline__ float dppmv(float x) {
    return __builtin_bit_cast(
        float, __builtin_amdgcn_update_dpp(__builtin_bit_cast(int, x),
                                           __builtin_bit_cast(int, x), CTRL,
                                           0xF, 0xF, true));
}
__device__ __forceinline__ float redmax16(float x) {
    x = fmaxf(x, dppmv<0xB1>(x));   // quad_perm [1,0,3,2]
    x = fmaxf(x, dppmv<0x4E>(x));   // quad_perm [2,3,0,1]
    x = fmaxf(x, dppmv<0x141>(x));  // row_half_mirror
    x = fmaxf(x, dppmv<0x140>(x));  // row_mirror (within 16)
    return x;
}
__device__ __forceinline__ float redsum16(float x) {
    x += dppmv<0xB1>(x);
    x += dppmv<0x4E>(x);
    x += dppmv<0x141>(x);
    x += dppmv<0x140>(x);
    return x;
}

// ---------------------------------------------------------------------------
// Weight prep: pack per-head wq/wk/wv [H,D,DH] fp32 into fused transposed
// Wqkv_t [3072][1024] bf16; q rows pre-scaled by 1/sqrt(DH)=0.125.
// ---------------------------------------------------------------------------
__global__ __launch_bounds__(256)
void make_wqkv_t(const float* __restrict__ wq, const float* __restrict__ wk,
                 const float* __restrict__ wv, bf16* __restrict__ out) {
    const int d = blockIdx.x * 256 + threadIdx.x;   // 0..1023
    const int n = blockIdx.y;                        // 0..3071
    const int which = n >> 10;
    const int hh = (n & 1023) >> 6;
    const int dh = n & 63;
    const float* src = (which == 0) ? wq : (which == 1) ? wk : wv;
    float v = src[((size_t)hh * 1024 + d) * 64 + dh];
    if (which == 0) v *= 0.125f;                     // 1/sqrt(64)
    out[(size_t)n * 1024 + d] = (bf16)v;
}

__global__ __launch_bounds__(256)
void transpose_to_bf16(const float* __restrict__ w, bf16* __restrict__ out) {
    const int d = blockIdx.x * 256 + threadIdx.x;    // 0..1023
    const int n = blockIdx.y;                        // 0..1023
    out[(size_t)n * 1024 + d] = (bf16)w[(size_t)d * 1024 + n];
}

// ---------------------------------------------------------------------------
// LayerNorm over D=1024; one block per row; optional bf16 and/or f32 outputs.
// ---------------------------------------------------------------------------
template <bool WB16, bool WF32>
__global__ __launch_bounds__(256)
void ln_kernel(const float* __restrict__ in, const float* __restrict__ gamma,
               const float* __restrict__ beta, bf16* __restrict__ ob,
               float* __restrict__ of) {
    const int row = blockIdx.x;
    const float* x = in + (size_t)row * 1024;
    float v[4];
    float s = 0.f, s2 = 0.f;
#pragma unroll
    for (int i = 0; i < 4; i++) {
        const int c = threadIdx.x + i * 256;
        v[i] = x[c];
        s += v[i];
        s2 += v[i] * v[i];
    }
    s = redsum16(s);
    s2 = redsum16(s2);
    s += __shfl_xor(s, 16, 32);
    s2 += __shfl_xor(s2, 16, 32);
    __shared__ float red[16];
    const int w = threadIdx.x >> 5;
    if ((threadIdx.x & 31) == 0) { red[w] = s; red[w + 8] = s2; }
    __syncthreads();
    s = 0.f; s2 = 0.f;
#pragma unroll
    for (int i = 0; i < 8; i++) { s += red[i]; s2 += red[i + 8]; }
    const float mean = s * (1.f / 1024.f);
    const float rstd = rsqrtf(s2 * (1.f / 1024.f) - mean * mean + 1e-5f);
#pragma unroll
    for (int i = 0; i < 4; i++) {
        const int c = threadIdx.x + i * 256;
        const float y = (v[i] - mean) * rstd * gamma[c] + beta[c];
        if (WB16) ob[(size_t)row * 1024 + c] = (bf16)y;
        if (WF32) of[(size_t)row * 1024 + c] = y;
    }
}

// ---------------------------------------------------------------------------
// Tiled WMMA GEMM: C[M,N] = A[M,K](bf16) * Bt[N,K](bf16)^T, block tile
// 128x256, BK=32, 8 waves x (64x64 wave tile) = 16 WMMAs per 8 fragment
// reads. Register-staged DOUBLE-BUFFERED LDS: next tile's global loads are
// issued before the WMMA block and committed to the alternate buffer after,
// so global latency hides behind the matrix ops; one barrier per k-step.
// ---------------------------------------------------------------------------
template <bool BF16OUT, bool RELU>
__global__ __launch_bounds__(256)
void gemm_wmma(const bf16* __restrict__ A, const bf16* __restrict__ Bt,
               const float* __restrict__ resid, float* __restrict__ Cf,
               bf16* __restrict__ Cb, int M, int N, int K) {
    constexpr int BM = 128, BN = 256, BK = 32, LDT = 48;
    __shared__ __align__(16) bf16 As[2][BM * LDT];   // 2 x 12 KB
    __shared__ __align__(16) bf16 Bs[2][BN * LDT];   // 2 x 24 KB

    const int tid = threadIdx.x;
    const int lane = tid & 31;
    const int wv = tid >> 5;
    const int half16 = (lane >> 4) & 1;
    const int l15 = lane & 15;

    const int m0 = blockIdx.y * BM;
    const int n0 = blockIdx.x * BN;
    const int wm = (wv >> 2) * 64;
    const int wn = (wv & 3) * 64;

    // per-thread cooperative-load coordinates (16B chunks)
    int arow[2], aoff[2], brow[4], boff[4];
#pragma unroll
    for (int c = 0; c < 2; c++) {
        const int cid = tid * 2 + c;
        arow[c] = cid >> 2;
        aoff[c] = (cid & 3) * 8;
    }
#pragma unroll
    for (int c = 0; c < 4; c++) {
        const int cid = tid * 4 + c;
        brow[c] = cid >> 2;
        boff[c] = (cid & 3) * 8;
    }

    v8f acc[4][4];
#pragma unroll
    for (int i = 0; i < 4; i++)
#pragma unroll
        for (int j = 0; j < 4; j++) acc[i][j] = vzero8f();

    // prologue: stage tile 0
    v8bf aR[2], bR[4];
#pragma unroll
    for (int c = 0; c < 2; c++)
        aR[c] = *(const v8bf*)&A[(size_t)(m0 + arow[c]) * K + aoff[c]];
#pragma unroll
    for (int c = 0; c < 4; c++)
        bR[c] = *(const v8bf*)&Bt[(size_t)(n0 + brow[c]) * K + boff[c]];
#pragma unroll
    for (int c = 0; c < 2; c++) *(v8bf*)&As[0][arow[c] * LDT + aoff[c]] = aR[c];
#pragma unroll
    for (int c = 0; c < 4; c++) *(v8bf*)&Bs[0][brow[c] * LDT + boff[c]] = bR[c];
    __syncthreads();

    int buf = 0;
    for (int kt = 0; kt < K; kt += BK, buf ^= 1) {
        const bool more = (kt + BK) < K;
        // prefetch next tile into registers (issues ahead of the WMMAs)
        if (more) {
#pragma unroll
            for (int c = 0; c < 2; c++)
                aR[c] = *(const v8bf*)&A[(size_t)(m0 + arow[c]) * K + kt + BK +
                                         aoff[c]];
#pragma unroll
            for (int c = 0; c < 4; c++)
                bR[c] = *(const v8bf*)&Bt[(size_t)(n0 + brow[c]) * K + kt + BK +
                                          boff[c]];
        }

        FragA af[4], bfr[4];
#pragma unroll
        for (int mi = 0; mi < 4; mi++) {
            const int row = wm + mi * 16 + l15;
            const int kb = half16 ? 8 : 0;  // A layout: K chunks {kb, kb+16}
            af[mi].h[0] = *(const v8bf*)&As[buf][row * LDT + kb];
            af[mi].h[1] = *(const v8bf*)&As[buf][row * LDT + kb + 16];
        }
#pragma unroll
        for (int ni = 0; ni < 4; ni++) {
            const int col = wn + ni * 16 + l15;
            const int kb = half16 ? 8 : 0;  // B layout: K chunks {kb, kb+16}
            bfr[ni].h[0] = *(const v8bf*)&Bs[buf][col * LDT + kb];
            bfr[ni].h[1] = *(const v8bf*)&Bs[buf][col * LDT + kb + 16];
        }
#pragma unroll
        for (int mi = 0; mi < 4; mi++)
#pragma unroll
            for (int ni = 0; ni < 4; ni++)
                acc[mi][ni] = wmma_bf16(af[mi].v, bfr[ni].v, acc[mi][ni]);

        // commit staged registers to the other buffer
        if (more) {
#pragma unroll
            for (int c = 0; c < 2; c++)
                *(v8bf*)&As[buf ^ 1][arow[c] * LDT + aoff[c]] = aR[c];
#pragma unroll
            for (int c = 0; c < 4; c++)
                *(v8bf*)&Bs[buf ^ 1][brow[c] * LDT + boff[c]] = bR[c];
        }
        __syncthreads();
    }

    // epilogue: C layout VGPR r -> row r + 8*half16, col = lane&15
#pragma unroll
    for (int mi = 0; mi < 4; mi++) {
#pragma unroll
        for (int ni = 0; ni < 4; ni++) {
            const int n = n0 + wn + ni * 16 + l15;
#pragma unroll
            for (int r = 0; r < 8; r++) {
                const int m = m0 + wm + mi * 16 + r + half16 * 8;
                float val = acc[mi][ni][r];
                if (RELU) val = val > 0.f ? val : 0.f;
                const size_t idx = (size_t)m * N + n;
                if (BF16OUT) {
                    Cb[idx] = (bf16)val;
                } else {
                    if (resid) val += resid[idx];
                    Cf[idx] = val;
                }
            }
        }
    }
}

// ---------------------------------------------------------------------------
// Flash attention: grid (S/128, B*H). 8 waves/block, each wave owns 16 query
// rows -> online-softmax row stats stay wave-private. All 8 K fragments for
// a key tile are hoisted into registers first (one long load clause with
// staged loadcnt waits feeding 8 back-to-back WMMAs). V tile staged
// transposed in double-buffered LDS with register prefetch overlapped with
// softmax + P@V. Softmax row max/sum use DPP16 butterflies (pure VALU).
// qkv layout: [B*S, 3072] bf16 (q pre-scaled | k | v), head-major (h*64+dh).
// ---------------------------------------------------------------------------
__global__ __launch_bounds__(256)
void attention_kernel(const bf16* __restrict__ qkv, bf16* __restrict__ concat) {
    constexpr int S = 2048, LDQ = 3072;
    const int bh = blockIdx.y;
    const int b = bh >> 4;
    const int h = bh & 15;
    const int wv = threadIdx.x >> 5;
    const int lane = threadIdx.x & 31;
    const int half16 = (lane >> 4) & 1;
    const int l15 = lane & 15;
    const int q0 = blockIdx.x * 128 + wv * 16;

    __shared__ __align__(16) bf16 Vt[2][64 * 72];    // [dh][t], stride 72
    __shared__ __align__(16) bf16 Pst[8][16 * 72];   // per-wave P staging

    const bf16* qbase = qkv + (size_t)(b * S) * LDQ + h * 64;
    const bf16* kbase = qbase + 1024;
    const bf16* vbase = qbase + 2048;

    // Q fragments: 2 K-chunks of 32 over dh
    FragA qf[2];
    {
        const bf16* qrow = qbase + (size_t)(q0 + l15) * LDQ;
#pragma unroll
        for (int kq = 0; kq < 2; kq++) {
            const int kb = kq * 32 + (half16 ? 8 : 0);
            qf[kq].h[0] = *(const v8bf*)(qrow + kb);
            qf[kq].h[1] = *(const v8bf*)(qrow + kb + 16);
        }
    }

    float mrow[8], lrow[8];
#pragma unroll
    for (int r = 0; r < 8; r++) { mrow[r] = -1e30f; lrow[r] = 0.f; }
    v8f cacc[4];
#pragma unroll
    for (int i = 0; i < 4; i++) cacc[i] = vzero8f();

    // V staging coordinates + prologue load of tile 0
    const int vt_t = threadIdx.x >> 2;            // 0..63
    const int vt_dhc = (threadIdx.x & 3) * 16;    // 0,16,32,48
    v8bf vr0 = *(const v8bf*)(vbase + (size_t)vt_t * LDQ + vt_dhc);
    v8bf vr1 = *(const v8bf*)(vbase + (size_t)vt_t * LDQ + vt_dhc + 8);

    int buf = 0;
    for (int t0 = 0; t0 < S; t0 += 64, buf ^= 1) {
        // commit staged V tile (transposed) into LDS buffer `buf`
#pragma unroll
        for (int i = 0; i < 8; i++) Vt[buf][(vt_dhc + i) * 72 + vt_t] = vr0[i];
#pragma unroll
        for (int i = 0; i < 8; i++)
            Vt[buf][(vt_dhc + 8 + i) * 72 + vt_t] = vr1[i];
        __syncthreads();

        // hoist ALL K fragments for this key tile -> one long load clause
        FragA kf[2][4];
#pragma unroll
        for (int ks = 0; ks < 2; ks++) {
#pragma unroll
            for (int nt = 0; nt < 4; nt++) {
                const bf16* kp = kbase +
                    (size_t)(t0 + nt * 16 + l15) * LDQ + ks * 32 +
                    (half16 ? 8 : 0);
                kf[ks][nt].h[0] = *(const v8bf*)(kp);
                kf[ks][nt].h[1] = *(const v8bf*)(kp + 16);
            }
        }

        // S = Q @ K^T  (16 rows x 64 keys per wave), 8 back-to-back WMMAs
        v8f sacc[4];
#pragma unroll
        for (int i = 0; i < 4; i++) sacc[i] = vzero8f();
#pragma unroll
        for (int ks = 0; ks < 2; ks++)
#pragma unroll
            for (int nt = 0; nt < 4; nt++)
                sacc[nt] = wmma_bf16(qf[ks].v, kf[ks][nt].v, sacc[nt]);

        // prefetch next V tile into registers (overlaps softmax + P@V)
        if (t0 + 64 < S) {
            const bf16* vp = vbase + (size_t)(t0 + 64 + vt_t) * LDQ + vt_dhc;
            vr0 = *(const v8bf*)(vp);
            vr1 = *(const v8bf*)(vp + 8);
        }

        // --- online softmax with DPP16 reductions ---
        float alpha[8];
#pragma unroll
        for (int r = 0; r < 8; r++) {
            float mx = fmaxf(fmaxf(sacc[0][r], sacc[1][r]),
                             fmaxf(sacc[2][r], sacc[3][r]));
            mx = redmax16(mx);
            const float mnew = fmaxf(mrow[r], mx);
            alpha[r] = __expf(mrow[r] - mnew);
            mrow[r] = mnew;
            float rs = 0.f;
#pragma unroll
            for (int nt = 0; nt < 4; nt++) {
                const float p = __expf(sacc[nt][r] - mnew);
                sacc[nt][r] = p;
                rs += p;
            }
            rs = redsum16(rs);
            lrow[r] = lrow[r] * alpha[r] + rs;
#pragma unroll
            for (int nt = 0; nt < 4; nt++) cacc[nt][r] *= alpha[r];
        }

        // P: C-layout -> LDS -> A-layout (wave-private, no barrier needed)
        bf16* pw = &Pst[wv][0];
#pragma unroll
        for (int nt = 0; nt < 4; nt++) {
            const int col = nt * 16 + l15;
#pragma unroll
            for (int r = 0; r < 8; r++)
                pw[(r + half16 * 8) * 72 + col] = (bf16)sacc[nt][r];
        }
        FragA pf[2];
#pragma unroll
        for (int ks = 0; ks < 2; ks++) {
            const int kb = ks * 32 + (half16 ? 8 : 0);
            pf[ks].h[0] = *(const v8bf*)(pw + l15 * 72 + kb);
            pf[ks].h[1] = *(const v8bf*)(pw + l15 * 72 + kb + 16);
        }

        // ctx += P @ V
#pragma unroll
        for (int ks = 0; ks < 2; ks++) {
#pragma unroll
            for (int nt = 0; nt < 4; nt++) {
                FragA vf;
                const bf16* vp2 =
                    &Vt[buf][(nt * 16 + l15) * 72 + ks * 32 +
                             (half16 ? 8 : 0)];
                vf.h[0] = *(const v8bf*)(vp2);
                vf.h[1] = *(const v8bf*)(vp2 + 16);
                cacc[nt] = wmma_bf16(pf[ks].v, vf.v, cacc[nt]);
            }
        }
        // no trailing barrier: next iteration writes the other V buffer
    }

    // finalize: ctx /= l, write head-concat layout [B*S, 1024] bf16
#pragma unroll
    for (int r = 0; r < 8; r++) {
        const float inv = 1.f / lrow[r];
        const int s = q0 + r + half16 * 8;
        const size_t rowbase = (size_t)(b * S + s) * 1024 + h * 64;
#pragma unroll
        for (int nt = 0; nt < 4; nt++)
            concat[rowbase + nt * 16 + l15] = (bf16)(cacc[nt][r] * inv);
    }
}

// ---------------------------------------------------------------------------
extern "C" void kernel_launch(void* const* d_in, const int* in_sizes, int n_in,
                              void* d_out, int out_size, void* d_ws,
                              size_t ws_size, hipStream_t stream) {
    const float* x    = (const float*)d_in[0];
    const float* wq   = (const float*)d_in[1];
    const float* wk   = (const float*)d_in[2];
    const float* wv   = (const float*)d_in[3];
    const float* wo   = (const float*)d_in[4];
    const float* w1   = (const float*)d_in[5];
    const float* w2   = (const float*)d_in[6];
    const float* ln1g = (const float*)d_in[7];
    const float* ln1b = (const float*)d_in[8];
    const float* ln2g = (const float*)d_in[9];
    const float* ln2b = (const float*)d_in[10];
    const float* ln3g = (const float*)d_in[11];
    const float* ln3b = (const float*)d_in[12];
    float* out = (float*)d_out;

    const size_t MB = 1024ull * 1024ull;
    char* ws = (char*)d_ws;
    bf16*  h1     = (bf16*)(ws + 0 * MB);    // [4096,1024] bf16 (8 MB)
    bf16*  qkv    = (bf16*)(ws + 8 * MB);    // [4096,3072] bf16 (24 MB)
    bf16*  concat = (bf16*)(ws + 32 * MB);   // [4096,1024] bf16 (8 MB)
    float* tmp    = (float*)(ws + 40 * MB);  // [4096,1024] f32 (16 MB)
    bf16*  h2b    = (bf16*)(ws + 56 * MB);   // [4096,1024] bf16 (8 MB)
    float* h2f    = (float*)(ws + 64 * MB);  // [4096,1024] f32 (16 MB)
    bf16*  wqkvt  = (bf16*)(ws + 80 * MB);   // [3072,1024] bf16 (6 MB)
    bf16*  wot    = (bf16*)(ws + 86 * MB);   // 2 MB
    bf16*  w1t    = (bf16*)(ws + 88 * MB);   // 2 MB
    bf16*  w2t    = (bf16*)(ws + 90 * MB);   // 2 MB
    bf16*  ffn1   = h1;                      // h1 is dead after QKV GEMM

    const dim3 b256(256);

    // weight prep (bf16, transposed; q pre-scaled by 1/8)
    make_wqkv_t<<<dim3(4, 3072), b256, 0, stream>>>(wq, wk, wv, wqkvt);
    transpose_to_bf16<<<dim3(4, 1024), b256, 0, stream>>>(wo, wot);
    transpose_to_bf16<<<dim3(4, 1024), b256, 0, stream>>>(w1, w1t);
    transpose_to_bf16<<<dim3(4, 1024), b256, 0, stream>>>(w2, w2t);

    // LN1 -> bf16
    ln_kernel<true, false><<<4096, b256, 0, stream>>>(x, ln1g, ln1b, h1, nullptr);

    // fused QKV projection: [4096,1024] x [1024,3072]
    gemm_wmma<true, false><<<dim3(12, 32), b256, 0, stream>>>(
        h1, wqkvt, nullptr, nullptr, qkv, 4096, 3072, 1024);

    // flash attention -> head-concat bf16
    attention_kernel<<<dim3(16, 32), b256, 0, stream>>>(qkv, concat);

    // output projection + residual(x) -> f32
    gemm_wmma<false, false><<<dim3(4, 32), b256, 0, stream>>>(
        concat, wot, x, tmp, nullptr, 4096, 1024, 1024);

    // LN2 -> bf16 (GEMM input) + f32 (FFN residual)
    ln_kernel<true, true><<<4096, b256, 0, stream>>>(tmp, ln2g, ln2b, h2b, h2f);

    // FFN1 + ReLU -> bf16
    gemm_wmma<true, true><<<dim3(4, 32), b256, 0, stream>>>(
        h2b, w1t, nullptr, nullptr, ffn1, 4096, 1024, 1024);

    // FFN2 + residual(h2) -> f32
    gemm_wmma<false, false><<<dim3(4, 32), b256, 0, stream>>>(
        ffn1, w2t, h2f, tmp, nullptr, 4096, 1024, 1024);

    // LN3 -> f32 output
    ln_kernel<false, true><<<4096, b256, 0, stream>>>(tmp, ln3g, ln3b, nullptr, out);
}
